// LatentVoxelGrid_38491496906992
// MI455X (gfx1250) — compile-verified, matrix operand order checked by reading, not run on hardware
//
#include <hip/hip_runtime.h>
#include <hip/hip_bf16.h>
#include <math.h>

typedef __attribute__((ext_vector_type(16))) _Float16 v16h;
typedef __attribute__((ext_vector_type(8)))  float    v8f;

#define NPTS 120000
#define KC   8
#define DD   64
#define MM   100000
#define PP   16
#define HG   32
#define HD   96

#define TAU_F   0.3f
#define OCC_F   0.5f
#define LMIN_F -2.0f
#define LMAX_F  3.5f

// ---------------------------------------------------------------------------
// WMMA helpers (CDNA5 wave32, D = A(16x32 f16) x B(32x16 f16) + C(16x16 f32))
// ---------------------------------------------------------------------------
static __device__ inline v8f wmma16(v16h a, v16h b, v8f c) {
  return __builtin_amdgcn_wmma_f32_16x16x32_f16(
      /*neg_a=*/false, a, /*neg_b=*/false, b,
      /*c_mod=*/(short)0, c, /*reuse_a=*/false, /*reuse_b=*/false);
}

// A operand from pre-swizzled LDS chunk [32 lanes][16 halfs]: contiguous 32B.
static __device__ inline v16h loadAs(const _Float16* chunk, int lane) {
  return *(const v16h*)(chunk + lane * 16);
}

// B operand: pre-packed per-lane-contiguous (32 bytes/lane) in global scratch.
static __device__ inline v16h loadB(const _Float16* p, int NT, int ck, int nt, int lane) {
  return *(const v16h*)(p + (((size_t)(ck * NT + nt) * 32 + lane) * 16));
}

// Swizzle (row, col) of a 16xK f16 tile into the WMMA-A per-lane layout:
// lane<16 : row=lane,    K = {0..7, 16..23} (+ck*32)
// lane>=16: row=lane-16, K = {8..15, 24..31} (+ck*32)
// Returns half-index within a [32][16] chunk; chunk index is col>>5.
static __device__ inline int aswz(int row, int col) {
  const int c32 = col & 31;
  const int g   = (c32 >> 3) & 1;
  const int j   = (c32 & 7) + ((c32 >> 4) << 3);
  return ((g << 4) | row) * 16 + j;
}

static __device__ inline float sigf(float x) { return 1.0f / (1.0f + __expf(-x)); }

// ---------------------------------------------------------------------------
// init: zero msg accumulator + wsum, seed vals into out[M..2M)
// ---------------------------------------------------------------------------
__global__ void k_init(const float* __restrict__ vals, float* __restrict__ out,
                       float* __restrict__ msg, float* __restrict__ wsum) {
  const int i = blockIdx.x * 256 + threadIdx.x;
  if (i < MM * DD) msg[i] = 0.0f;
  if (i < MM) { wsum[i] = 0.0f; out[MM + i] = vals[i]; }
}

// ---------------------------------------------------------------------------
// pack weight matrix W (Nrows x Kdim, row-major f32) into WMMA-B f16 layout
// ---------------------------------------------------------------------------
__global__ void k_pack(const float* __restrict__ W, _Float16* __restrict__ dst,
                       int Nrows, int Kdim) {
  const int NT = Nrows >> 4;
  const int total = Nrows * Kdim;
  const int tid = blockIdx.x * 256 + threadIdx.x;
  if (tid >= total) return;
  const int j    = tid & 15;
  const int lane = (tid >> 4) & 31;
  const int tile = tid >> 9;
  const int ck = tile / NT;
  const int nt = tile % NT;
  const int r  = nt * 16 + (lane & 15);
  const int g  = lane >> 4;
  const int kk = ck * 32 + (j & 7) + 8 * g + 16 * (j >> 3);
  dst[tid] = (_Float16)W[r * Kdim + kk];
}

// ---------------------------------------------------------------------------
// per-point: hash+searchsorted scatter, routing softmax, weighted scatter
// ---------------------------------------------------------------------------
__global__ __launch_bounds__(256) void k_points(
    const float* __restrict__ pts, const float* __restrict__ f_pts,
    const float* __restrict__ z_lat, const float* __restrict__ centers,
    const float* __restrict__ Wf, const float* __restrict__ Wz,
    const float* __restrict__ w_delta, const float* __restrict__ b_delta,
    const float* __restrict__ log_temp,
    const int* __restrict__ keys, const int* __restrict__ cand,
    float* __restrict__ out_vals, float* __restrict__ msg,
    float* __restrict__ wsum) {
  __shared__ float sWf[PP * DD];
  __shared__ float sWz[PP * DD];
  for (int i = threadIdx.x; i < PP * DD; i += 256) { sWf[i] = Wf[i]; sWz[i] = Wz[i]; }
  __syncthreads();

  const int n = blockIdx.x * 256 + threadIdx.x;
  if (n >= NPTS) return;

  const float px = pts[3 * n], py = pts[3 * n + 1], pz = pts[3 * n + 2];
  const int ix = (int)floorf(px / 0.1f);
  const int iy = (int)floorf(py / 0.1f);
  const int iz = (int)floorf(pz / 0.1f);
  const int h = (int)((((unsigned)(ix + 512)) << 20) ^
                      (((unsigned)(iy + 512)) << 10) ^
                      ((unsigned)(iz + 512)));
  int lo = 0, hi = MM;
  while (lo < hi) { int mid = (lo + hi) >> 1; if (keys[mid] < h) lo = mid + 1; else hi = mid; }
  const int vox = lo < MM ? lo : MM - 1;
  atomicAdd(&out_vals[vox], OCC_F);

  const float* frow = f_pts + (size_t)n * DD;

  // Fp = f . Wf^T  (16 projections)
  float Fp[PP];
#pragma unroll
  for (int p = 0; p < PP; ++p) Fp[p] = 0.0f;
  for (int d = 0; d < DD; ++d) {
    const float fv = frow[d];
#pragma unroll
    for (int p = 0; p < PP; ++p) Fp[p] += sWf[p * DD + d] * fv;
  }
  float nf = 0.0f;
#pragma unroll
  for (int p = 0; p < PP; ++p) nf += Fp[p] * Fp[p];
  nf = sqrtf(nf) + 1e-6f;

  const float wd0 = w_delta[0], wd1 = w_delta[1], wd2 = w_delta[2];
  const float bdl = b_delta[0];
  const float expT = __expf(log_temp[0]);

  int idxs[KC];
  float sim[KC];
#pragma unroll
  for (int k = 0; k < KC; ++k) {
    const int idx = cand[n * KC + k];
    idxs[k] = idx;
    __builtin_prefetch(z_lat + (size_t)idx * DD, 0, 1);
  }
#pragma unroll
  for (int k = 0; k < KC; ++k) {
    const int idx = idxs[k];
    const float* zrow = z_lat + (size_t)idx * DD;
    float Zp[PP];
#pragma unroll
    for (int p = 0; p < PP; ++p) Zp[p] = 0.0f;
    for (int d = 0; d < DD; ++d) {
      const float zv = zrow[d];
#pragma unroll
      for (int p = 0; p < PP; ++p) Zp[p] += sWz[p * DD + d] * zv;
    }
    float nz = 0.0f, dot = 0.0f;
#pragma unroll
    for (int p = 0; p < PP; ++p) { nz += Zp[p] * Zp[p]; dot += Fp[p] * Zp[p]; }
    nz = sqrtf(nz) + 1e-6f;
    const float core = dot / (nf * nz);
    const float dx = px - centers[3 * idx];
    const float dy = py - centers[3 * idx + 1];
    const float dz = pz - centers[3 * idx + 2];
    sim[k] = expT * (core + dx * wd0 + dy * wd1 + dz * wd2 + bdl);
  }

  float mx = sim[0];
#pragma unroll
  for (int k = 1; k < KC; ++k) mx = fmaxf(mx, sim[k]);
  float wv[KC], s = 0.0f;
#pragma unroll
  for (int k = 0; k < KC; ++k) { wv[k] = __expf((sim[k] - mx) / TAU_F); s += wv[k]; }
  const float inv = 1.0f / s;
#pragma unroll
  for (int k = 0; k < KC; ++k) {
    const float wk = wv[k] * inv;
    const int idx = idxs[k];
    atomicAdd(&wsum[idx], wk);
    float* mrow = msg + (size_t)idx * DD;
    for (int d = 0; d < DD; ++d) atomicAdd(&mrow[d], wk * frow[d]);
  }
}

__global__ void k_clip(float* __restrict__ out_vals) {
  const int i = blockIdx.x * 256 + threadIdx.x;
  if (i < MM) out_vals[i] = fminf(fmaxf(out_vals[i], LMIN_F), LMAX_F);
}

// ---------------------------------------------------------------------------
// fused per-voxel: gate MLP + GRU + LayerNorm + decoder, WMMA throughout.
// 64 threads/block = 2 waves, each wave owns a 16-row voxel tile.
// msg tile is fetched with GLOBAL_LOAD_ASYNC_TO_LDS (ASYNCcnt path), A
// operands live in pre-swizzled LDS chunks so each fetch is one 32B DS read.
// ---------------------------------------------------------------------------
__global__ __launch_bounds__(64) void k_voxel(
    const float* __restrict__ z_lat, const float* __restrict__ msg_acc,
    const float* __restrict__ wsum,
    const _Float16* __restrict__ Wihp, const _Float16* __restrict__ Whhp,
    const _Float16* __restrict__ Wg1p, const _Float16* __restrict__ Wd1p,
    const _Float16* __restrict__ Wd2p,
    const float* __restrict__ b_ih, const float* __restrict__ b_hh,
    const float* __restrict__ bg1, const float* __restrict__ Wg2,
    const float* __restrict__ bg2,
    const float* __restrict__ ln_g, const float* __restrict__ ln_b,
    const float* __restrict__ bd1, const float* __restrict__ bd2,
    const float* __restrict__ Wd3, const float* __restrict__ bd3,
    float* __restrict__ out_p) {
  __shared__ float mraw[2][16][DD];                  // async-landed raw msg sums
  __shared__ float zf[2][16][DD];                    // z, overwritten with z_new
  __shared__ __align__(32) _Float16 zhs[2][2][32 * 16];  // z  WMMA-A swizzled
  __shared__ __align__(32) _Float16 mhs[2][2][32 * 16];  // msg (then x) swizzled
  __shared__ __align__(32) _Float16 hhs[2][3][32 * 16];  // hd swizzled (K=96)
  __shared__ float gld[2][16];                       // gate g per row
  __shared__ float hf[2][16][HD];                    // hd f32 (then hd2)

  const int w    = threadIdx.x >> 5;
  const int lane = threadIdx.x & 31;
  const int r0   = blockIdx.x * 32 + w * 16;
  const int c    = lane & 15;
  const int g2   = lane >> 4;

  // ---- async copy raw msg tile: global -> LDS, 16B per lane per issue ----
  for (int i = lane; i < 16 * 16; i += 32) {         // 16 rows x 16 xb128 slots
    const int row = i >> 4, q = i & 15;
    int gr = r0 + row; if (gr >= MM) gr = MM - 1;
    const float* src = msg_acc + (size_t)gr * DD + q * 4;
    const unsigned ldsoff = (unsigned)(uintptr_t)(&mraw[w][row][q * 4]);
    asm volatile("global_load_async_to_lds_b128 %0, %1, off"
                 :: "v"(ldsoff), "v"((unsigned long long)(uintptr_t)src)
                 : "memory");
  }

  // ---- load z (VGPR path) while async msg copy is in flight ----
  for (int i = lane; i < 16 * DD; i += 32) {
    const int row = i >> 6, col = i & 63;
    int gr = r0 + row; if (gr >= MM) gr = MM - 1;
    const float zv = z_lat[(size_t)gr * DD + col];
    zf[w][row][col] = zv;
    zhs[w][col >> 5][aswz(row, col)] = (_Float16)zv;
  }

  asm volatile("s_wait_asynccnt 0" ::: "memory");
  __syncthreads();

  // ---- normalize msg and swizzle to f16 A layout ----
  for (int i = lane; i < 16 * DD; i += 32) {
    const int row = i >> 6, col = i & 63;
    int gr = r0 + row; if (gr >= MM) gr = MM - 1;
    const float mv = mraw[w][row][col] / (wsum[gr] + 1e-6f);
    mhs[w][col >> 5][aswz(row, col)] = (_Float16)mv;
  }
  __syncthreads();

  // ---- hoist A operands: one 32B DS read each, reused across all tiles ----
  const v16h az0 = loadAs(zhs[w][0], lane);
  const v16h az1 = loadAs(zhs[w][1], lane);
  const v16h am0 = loadAs(mhs[w][0], lane);
  const v16h am1 = loadAs(mhs[w][1], lane);

  // ---- gate: sigmoid(relu([z|msg] Wg1^T + bg1) Wg2^T + bg2) ----
  {
    v8f acc0 = {}, acc1 = {};
    acc0 = wmma16(az0, loadB(Wg1p, 2, 0, 0, lane), acc0);
    acc1 = wmma16(az0, loadB(Wg1p, 2, 0, 1, lane), acc1);
    acc0 = wmma16(az1, loadB(Wg1p, 2, 1, 0, lane), acc0);
    acc1 = wmma16(az1, loadB(Wg1p, 2, 1, 1, lane), acc1);
    acc0 = wmma16(am0, loadB(Wg1p, 2, 2, 0, lane), acc0);
    acc1 = wmma16(am0, loadB(Wg1p, 2, 2, 1, lane), acc1);
    acc0 = wmma16(am1, loadB(Wg1p, 2, 3, 0, lane), acc0);
    acc1 = wmma16(am1, loadB(Wg1p, 2, 3, 1, lane), acc1);
    const float b0 = bg1[c], b1 = bg1[16 + c];
    const float w20 = Wg2[c], w21 = Wg2[16 + c];
    float s[8];
#pragma unroll
    for (int v = 0; v < 8; ++v) {
      float t = fmaxf(acc0[v] + b0, 0.0f) * w20 + fmaxf(acc1[v] + b1, 0.0f) * w21;
#pragma unroll
      for (int m = 1; m < 16; m <<= 1) t += __shfl_xor(t, m, 16);
      s[v] = t;
    }
    if (c == 0) {
      const float bb = bg2[0];
#pragma unroll
      for (int v = 0; v < 8; ++v) gld[w][v + 8 * g2] = sigf(s[v] + bb);
    }
  }
  __syncthreads();

  // ---- GRU fusion, per 16-col tile of D; z_new written into zf in place ----
#pragma unroll
  for (int dt = 0; dt < 4; ++dt) {
    v8f aR = {}, aU = {}, aNi = {}, aNh = {};
    aR  = wmma16(am0, loadB(Wihp, 12, 0, dt,     lane), aR);
    aR  = wmma16(am1, loadB(Wihp, 12, 1, dt,     lane), aR);
    aR  = wmma16(az0, loadB(Whhp, 12, 0, dt,     lane), aR);
    aR  = wmma16(az1, loadB(Whhp, 12, 1, dt,     lane), aR);
    aU  = wmma16(am0, loadB(Wihp, 12, 0, 4 + dt, lane), aU);
    aU  = wmma16(am1, loadB(Wihp, 12, 1, 4 + dt, lane), aU);
    aU  = wmma16(az0, loadB(Whhp, 12, 0, 4 + dt, lane), aU);
    aU  = wmma16(az1, loadB(Whhp, 12, 1, 4 + dt, lane), aU);
    aNi = wmma16(am0, loadB(Wihp, 12, 0, 8 + dt, lane), aNi);
    aNi = wmma16(am1, loadB(Wihp, 12, 1, 8 + dt, lane), aNi);
    aNh = wmma16(az0, loadB(Whhp, 12, 0, 8 + dt, lane), aNh);
    aNh = wmma16(az1, loadB(Whhp, 12, 1, 8 + dt, lane), aNh);
    const int col = dt * 16 + c;
    const float bR  = b_ih[col] + b_hh[col];
    const float bU  = b_ih[64 + col] + b_hh[64 + col];
    const float bNi = b_ih[128 + col];
    const float bNh = b_hh[128 + col];
#pragma unroll
    for (int v = 0; v < 8; ++v) {
      const int row = v + 8 * g2;
      const float rg = sigf(aR[v] + bR);
      const float ug = sigf(aU[v] + bU);
      const float nn = tanhf(aNi[v] + bNi + rg * (aNh[v] + bNh));
      const float zv = zf[w][row][col];
      const float hn = (1.0f - ug) * nn + ug * zv;
      zf[w][row][col] = zv + gld[w][row] * (hn - zv);
    }
  }
  __syncthreads();

  // ---- LayerNorm on z_new -> x (swizzled f16, reuse mhs) ----
  if (lane < 16) {
    const int row = lane;
    float mu = 0.0f;
    for (int d = 0; d < DD; ++d) mu += zf[w][row][d];
    mu *= (1.0f / DD);
    float var = 0.0f;
    for (int d = 0; d < DD; ++d) { const float t = zf[w][row][d] - mu; var += t * t; }
    var *= (1.0f / DD);
    const float rs = rsqrtf(var + 1e-5f);
    for (int d = 0; d < DD; ++d) {
      const float xh = (zf[w][row][d] - mu) * rs * ln_g[d] + ln_b[d];
      mhs[w][d >> 5][aswz(row, d)] = (_Float16)xh;
    }
  }
  __syncthreads();

  // ---- fc1: hd = relu(x Wd1^T + bd1) ----
  {
    const v16h ax0 = loadAs(mhs[w][0], lane);
    const v16h ax1 = loadAs(mhs[w][1], lane);
#pragma unroll
    for (int nt = 0; nt < 6; ++nt) {
      v8f acc = {};
      acc = wmma16(ax0, loadB(Wd1p, 6, 0, nt, lane), acc);
      acc = wmma16(ax1, loadB(Wd1p, 6, 1, nt, lane), acc);
      const int col = nt * 16 + c;
      const float bv = bd1[col];
#pragma unroll
      for (int v = 0; v < 8; ++v) {
        const int row = v + 8 * g2;
        const float h1 = fmaxf(acc[v] + bv, 0.0f);
        hf[w][row][col] = h1;
        hhs[w][col >> 5][aswz(row, col)] = (_Float16)h1;
      }
    }
  }
  __syncthreads();

  // ---- fc2 residual: hd2 = hd + relu(hd Wd2^T + bd2) ----
  {
    const v16h ah0 = loadAs(hhs[w][0], lane);
    const v16h ah1 = loadAs(hhs[w][1], lane);
    const v16h ah2 = loadAs(hhs[w][2], lane);
#pragma unroll
    for (int nt = 0; nt < 6; ++nt) {
      v8f acc = {};
      acc = wmma16(ah0, loadB(Wd2p, 6, 0, nt, lane), acc);
      acc = wmma16(ah1, loadB(Wd2p, 6, 1, nt, lane), acc);
      acc = wmma16(ah2, loadB(Wd2p, 6, 2, nt, lane), acc);
      const int col = nt * 16 + c;
      const float bv = bd2[col];
#pragma unroll
      for (int v = 0; v < 8; ++v) {
        const int row = v + 8 * g2;
        hf[w][row][col] = hf[w][row][col] + fmaxf(acc[v] + bv, 0.0f);
      }
    }
  }
  __syncthreads();

  // ---- fc3 + sigmoid ----
  if (lane < 16) {
    const int gr = r0 + lane;
    if (gr < MM) {
      float acc = bd3[0];
      for (int d = 0; d < HD; ++d) acc += hf[w][lane][d] * Wd3[d];
      out_p[gr] = sigf(acc);
    }
  }
}

// ---------------------------------------------------------------------------
extern "C" void kernel_launch(void* const* d_in, const int* in_sizes, int n_in,
                              void* d_out, int out_size, void* d_ws, size_t ws_size,
                              hipStream_t stream) {
  const float* pts      = (const float*)d_in[0];
  const float* f_pts    = (const float*)d_in[1];
  const float* z_lat    = (const float*)d_in[2];
  const float* vals     = (const float*)d_in[3];
  const float* centers  = (const float*)d_in[4];
  const float* Wf       = (const float*)d_in[5];
  const float* Wz       = (const float*)d_in[6];
  const float* w_delta  = (const float*)d_in[7];
  const float* b_delta  = (const float*)d_in[8];
  const float* log_temp = (const float*)d_in[9];
  const float* W_ih     = (const float*)d_in[10];
  const float* W_hh     = (const float*)d_in[11];
  const float* b_ih     = (const float*)d_in[12];
  const float* b_hh     = (const float*)d_in[13];
  const float* Wg1      = (const float*)d_in[14];
  const float* bg1      = (const float*)d_in[15];
  const float* Wg2      = (const float*)d_in[16];
  const float* bg2      = (const float*)d_in[17];
  const float* ln_g     = (const float*)d_in[18];
  const float* ln_b     = (const float*)d_in[19];
  const float* Wd1      = (const float*)d_in[20];
  const float* bd1      = (const float*)d_in[21];
  const float* Wd2      = (const float*)d_in[22];
  const float* bd2      = (const float*)d_in[23];
  const float* Wd3      = (const float*)d_in[24];
  const float* bd3      = (const float*)d_in[25];
  const int*   keys     = (const int*)d_in[26];
  const int*   cand     = (const int*)d_in[27];
  float* out = (float*)d_out;

  char* ws = (char*)d_ws;
  float* msg   = (float*)ws;                       // M*D f32 = 25,600,000 B
  float* wsumb = (float*)(ws + 25600000);          // M f32   =    400,000 B
  _Float16* packBase = (_Float16*)(ws + 26000000); // packed f16 weights
  _Float16* Wihp = packBase;                       // 192x64 -> 12288 halfs
  _Float16* Whhp = packBase + 12288;               // 192x64
  _Float16* Wg1p = packBase + 24576;               // 32x128 -> 4096
  _Float16* Wd1p = packBase + 28672;               // 96x64  -> 6144
  _Float16* Wd2p = packBase + 34816;               // 96x96  -> 9216

  k_init<<<(MM * DD + 255) / 256, 256, 0, stream>>>(vals, out, msg, wsumb);

  k_pack<<<48, 256, 0, stream>>>(W_ih, Wihp, 192, 64);
  k_pack<<<48, 256, 0, stream>>>(W_hh, Whhp, 192, 64);
  k_pack<<<16, 256, 0, stream>>>(Wg1, Wg1p, 32, 128);
  k_pack<<<24, 256, 0, stream>>>(Wd1, Wd1p, 96, 64);
  k_pack<<<36, 256, 0, stream>>>(Wd2, Wd2p, 96, 96);

  k_points<<<(NPTS + 255) / 256, 256, 0, stream>>>(
      pts, f_pts, z_lat, centers, Wf, Wz, w_delta, b_delta, log_temp,
      keys, cand, out + MM, msg, wsumb);

  k_clip<<<(MM + 255) / 256, 256, 0, stream>>>(out + MM);

  k_voxel<<<(MM + 31) / 32, 64, 0, stream>>>(
      z_lat, msg, wsumb, Wihp, Whhp, Wg1p, Wd1p, Wd2p,
      b_ih, b_hh, bg1, Wg2, bg2, ln_g, ln_b, bd1, bd2, Wd3, bd3, out);
}
